// MusicAttrCVAE_14963666059401
// MI455X (gfx1250) — matrix operand order repaired and checked by reading, not compiled, not run on hardware
//
#include <hip/hip_runtime.h>
#include <hip/hip_bf16.h>
#include <stdint.h>

// ---------------- problem constants ----------------
#define BSZ   64
#define TSZ   256
#define ROLL  130
#define HID   2048
#define ZD    128
#define G3    (3 * HID)        // 6144 gate width
#define XENC_P 160             // ROLL+2=132 padded to mult of 32
#define XDEC_P 288             // ZD+2+ROLL=260 padded to mult of 32
#define HCAT  (2 * HID)        // 4096
#define NOUT_P 256             // ROLL padded for B-matrix reads

typedef __attribute__((ext_vector_type(16))) __bf16 v16bf;
typedef __attribute__((ext_vector_type(8)))  float  v8f;
typedef __attribute__((ext_vector_type(4)))  unsigned int u32x4;
typedef __attribute__((ext_vector_type(4)))  int  i32x4;
typedef __attribute__((ext_vector_type(8)))  int  i32x8;

#if defined(__has_builtin)
#if __has_builtin(__builtin_amdgcn_tensor_load_to_lds)
#define USE_TDM 1
#else
#define USE_TDM 0
#endif
#else
#define USE_TDM 0
#endif

#if USE_TDM
// Issue a 2D TDM tile load: rows x cols (bf16 elements) from gptr (row stride
// lda elements) into LDS at lds_byte_addr, packed contiguous [rows][cols].
// D# bitfields per CDNA5 ISA 8.3/8.4 (group0: count/lds/global/type,
// group1: data_size, tensor_dim0/1, tile_dim0/1, tensor_dim0_stride).
__device__ __forceinline__ void tdm_load_tile(const __bf16* gptr, int lda,
                                              unsigned int lds_byte_addr,
                                              int rows, int cols, int tensor_rows) {
    unsigned long long ga = (unsigned long long)(uintptr_t)gptr;
    u32x4 g0;
    g0[0] = 1u;                                          // count=1 (valid user D#)
    g0[1] = lds_byte_addr;                               // lds_addr
    g0[2] = (unsigned int)(ga & 0xFFFFFFFFu);            // global_addr[31:0]
    g0[3] = (unsigned int)((ga >> 32) & 0x01FFFFFFu)     // global_addr[56:32]
          | (2u << 30);                                  // type=2 ("image")
    unsigned int td0 = (unsigned int)lda;                // tensor_dim0 (row length)
    unsigned int td1 = (unsigned int)tensor_rows;        // tensor_dim1 (#rows)
    unsigned long long st0 = (unsigned long long)lda;    // tensor_dim0_stride
    i32x8 g1;
    g1[0] = (int)(1u << 16);                             // data_size=1 -> 2 bytes
    g1[1] = (int)((td0 & 0xFFFFu) << 16);                // tensor_dim0[15:0]
    g1[2] = (int)((td0 >> 16) | ((td1 & 0xFFFFu) << 16));// dim0[31:16] | dim1[15:0]
    g1[3] = (int)((td1 >> 16) | ((unsigned int)cols << 16)); // dim1[31:16] | tile_dim0
    g1[4] = (int)(unsigned int)rows;                     // tile_dim1 (tile_dim2=0)
    g1[5] = (int)(unsigned int)(st0 & 0xFFFFFFFFu);      // dim0_stride[31:0]
    g1[6] = (int)(unsigned int)((st0 >> 32) & 0xFFFFu);  // dim0_stride[47:32]
    g1[7] = 0;
    i32x4 z4 = {0, 0, 0, 0};
#if __clang_major__ >= 23
    i32x8 z8 = {0, 0, 0, 0, 0, 0, 0, 0};
    __builtin_amdgcn_tensor_load_to_lds(g0, g1, z4, z4, z8, 0);
#else
    __builtin_amdgcn_tensor_load_to_lds(g0, g1, z4, z4, 0);
#endif
}
#endif

// ---------------- WMMA bf16 GEMM with TDM-staged A tiles ----------------
// C[M,N](f32) = A[M,K](bf16, lda) x Bm[K,N](bf16, ldb, K-major) + bias.
// block = 256 threads = 8 waves; block tile = 64 (M) x 128 (N).
// Wave (wm,wn): wm=wave>>1 owns 16 M-rows, wn=wave&1 owns 64 N-cols = 4 WMMA tiles.
// Pipeline: TDM double-buffers the 64x32 A tile in LDS while B fragments are
// double-buffered in registers, so the 4 WMMAs per K-step overlap both the
// next A-tile DMA and the next B-fragment loads; one barrier per K-step.
__launch_bounds__(256)
__global__ void k_gemm_bf16(const __bf16* __restrict__ A, int lda,
                            const __bf16* __restrict__ Bm, int ldb,
                            const float* __restrict__ bias,
                            float* __restrict__ C, int ldc,
                            int M, int N, int K) {
    __shared__ __align__(128) __bf16 As[2][64][32];
    const int tid  = threadIdx.x;
    const int wave = tid >> 5;
    const int lane = tid & 31;
    const int wm   = wave >> 1;          // 0..3
    const int wn   = wave & 1;           // 0..1
    const int m0   = blockIdx.y * 64;
    const int n0w  = blockIdx.x * 128 + wn * 64;
    const int half = lane >> 4;
    const int r    = lane & 15;
    const int iters = K >> 5;

#if USE_TDM
    if (wave == 0) {
        tdm_load_tile(A + (size_t)m0 * lda, lda,
                      (unsigned int)(uintptr_t)&As[0][0][0], 64, 32, M);
    }
#else
    {   // cooperative fallback: 64x32 bf16 = 1024 dwords
        for (int e = tid * 2; e < 64 * 32; e += 512) {
            int ar = e >> 5, ac = e & 31;
            *(uint32_t*)&As[0][ar][ac] =
                *(const uint32_t*)(A + (size_t)(m0 + ar) * lda + ac);
        }
    }
#endif

    // strength-reduced B pointer: rows k0+16*half.., col n0w+r (+16j per tile)
    const __bf16* brow = Bm + (size_t)(half * 16) * ldb + n0w + r;
    const size_t  bstep = (size_t)32 * ldb;

    // preload K-step 0 B fragments into registers
    v16bf b0, b1, b2, b3;
    for (int s = 0; s < 16; ++s) {
        const __bf16* q = brow + (size_t)s * ldb;
        b0[s] = q[0]; b1[s] = q[16]; b2[s] = q[32]; b3[s] = q[48];
    }

    v8f acc0 = {}, acc1 = {}, acc2 = {}, acc3 = {};
    for (int it = 0; it < iters; ++it) {
        const int buf = it & 1;
#if USE_TDM
        if (wave == 0) __builtin_amdgcn_s_wait_tensorcnt(0);
#endif
        __syncthreads();   // staged A[buf] visible to all waves
        const bool more = (it + 1 < iters);
        if (more) {
#if USE_TDM
            if (wave == 0) {
                tdm_load_tile(A + (size_t)m0 * lda + ((it + 1) << 5), lda,
                              (unsigned int)(uintptr_t)&As[buf ^ 1][0][0], 64, 32, M);
            }
#else
            const int k1 = (it + 1) << 5;
            for (int e = tid * 2; e < 64 * 32; e += 512) {
                int ar = e >> 5, ac = e & 31;
                *(uint32_t*)&As[buf ^ 1][ar][ac] =
                    *(const uint32_t*)(A + (size_t)(m0 + ar) * lda + k1 + ac);
            }
#endif
        }
        // issue next K-step's B loads before the WMMAs (latency hidden by MFU)
        v16bf nb0 = b0, nb1 = b1, nb2 = b2, nb3 = b3;
        const __bf16* nbp = brow + bstep;
        if (more) {
            for (int s = 0; s < 16; ++s) {
                const __bf16* q = nbp + (size_t)s * ldb;
                nb0[s] = q[0]; nb1[s] = q[16]; nb2[s] = q[32]; nb3[s] = q[48];
            }
            if (it + 3 < iters) __builtin_prefetch(brow + 3 * bstep, 0, 1);
        }
        v16bf a = *(const v16bf*)&As[buf][wm * 16 + r][half * 16];
        acc0 = __builtin_amdgcn_wmma_f32_16x16x32_bf16(false, a, false, b0, (short)0, acc0, false, false);
        acc1 = __builtin_amdgcn_wmma_f32_16x16x32_bf16(false, a, false, b1, (short)0, acc1, false, false);
        acc2 = __builtin_amdgcn_wmma_f32_16x16x32_bf16(false, a, false, b2, (short)0, acc2, false, false);
        acc3 = __builtin_amdgcn_wmma_f32_16x16x32_bf16(false, a, false, b3, (short)0, acc3, false, false);
        b0 = nb0; b1 = nb1; b2 = nb2; b3 = nb3;
        brow = nbp;
#if !USE_TDM
        __syncthreads();   // fallback: writers/readers of As separated
#endif
    }

    const int rowbase = m0 + wm * 16 + half * 8;
    v8f accs[4] = {acc0, acc1, acc2, acc3};
    #pragma unroll
    for (int j = 0; j < 4; ++j) {
        const int col = n0w + j * 16 + r;
        if (col < N) {
            const float bv = bias ? bias[col] : 0.0f;
            #pragma unroll
            for (int i = 0; i < 8; ++i) {
                const int row = rowbase + i;
                if (row < M) C[(size_t)row * ldc + col] = accs[j][i] + bv;
            }
        }
    }
}

// ---------------- weight transpose + cast ----------------
__global__ void k_transpose_cast(const float* __restrict__ W, int N, int K,
                                 __bf16* __restrict__ out, int KP, int NP) {
    int idx = blockIdx.x * 256 + threadIdx.x;
    if (idx >= KP * NP) return;
    int np = idx % NP;
    int kp = idx / NP;
    float v = (kp < K && np < N) ? W[(size_t)np * K + kp] : 0.0f;
    out[idx] = (__bf16)v;
}

// ---------------- GRU gate fusion ----------------
__global__ void k_gates(const float* __restrict__ gi, const float* __restrict__ gh,
                        const float* __restrict__ hprev, float* __restrict__ hout,
                        __bf16* __restrict__ act, __bf16* __restrict__ h_all_t) {
    int idx = blockIdx.x * 256 + threadIdx.x;     // 64*2048 threads
    int b = idx >> 11;
    int j = idx & (HID - 1);
    const float* gib = gi + (size_t)b * G3;
    const float* ghb = gh + (size_t)b * G3;
    float ir = gib[j], iz = gib[HID + j], inn = gib[2 * HID + j];
    float hr = ghb[j], hz = ghb[HID + j], hn  = ghb[2 * HID + j];
    float rg = 1.0f / (1.0f + __expf(-(ir + hr)));
    float zg = 1.0f / (1.0f + __expf(-(iz + hz)));
    float n  = tanhf(inn + rg * hn);
    float h  = hprev[idx];
    float hnew = (1.0f - zg) * n + zg * h;
    hout[idx] = hnew;
    act[idx]  = (__bf16)hnew;
    if (h_all_t) h_all_t[idx] = (__bf16)hnew;
}

// ---------------- encoder input packing ----------------
__global__ void k_prep_enc(const float* __restrict__ x, const float* __restrict__ rd,
                           const float* __restrict__ nd,
                           __bf16* __restrict__ xf, __bf16* __restrict__ xb) {
    int idx = blockIdx.x * 256 + threadIdx.x;
    if (idx >= TSZ * BSZ * XENC_P) return;
    int c = idx % XENC_P;
    int b = (idx / XENC_P) % BSZ;
    int t = idx / (XENC_P * BSZ);
    float vf, vb;
    if (c < ROLL) {
        vf = x[((size_t)b * TSZ + t) * ROLL + c];
        vb = x[((size_t)b * TSZ + (TSZ - 1 - t)) * ROLL + c];
    } else if (c == ROLL)     { vf = vb = rd[b]; }
    else if (c == ROLL + 1)   { vf = vb = nd[b]; }
    else                      { vf = vb = 0.0f; }
    xf[idx] = (__bf16)vf;
    xb[idx] = (__bf16)vb;
}

// ---------------- hcat = [h_f, h_b] -> bf16 ----------------
__global__ void k_concat_cast(const float* __restrict__ hf, const float* __restrict__ hb,
                              __bf16* __restrict__ out) {
    int idx = blockIdx.x * 256 + threadIdx.x;   // 64*4096
    int b = idx >> 12;
    int c = idx & (HCAT - 1);
    float v = (c < HID) ? hf[(size_t)b * HID + c] : hb[(size_t)b * HID + (c - HID)];
    out[idx] = (__bf16)v;
}

// ---------------- reparameterize + outputs mu/std/z ----------------
__global__ void k_finalize_z(const float* __restrict__ gm, const float* __restrict__ gv,
                             const float* __restrict__ eps,
                             const float* __restrict__ rd, const float* __restrict__ nd,
                             float* __restrict__ d_mu, float* __restrict__ d_std,
                             float* __restrict__ d_z,
                             float* __restrict__ zf, __bf16* __restrict__ z16) {
    int idx = blockIdx.x * 256 + threadIdx.x;   // 64*160
    if (idx >= BSZ * XENC_P) return;
    int b = idx / XENC_P;
    int c = idx % XENC_P;
    if (c < ZD) {
        float mu  = gm[(size_t)b * ZD + c];
        float st  = __expf(gv[(size_t)b * ZD + c]);
        float z   = mu + st * eps[(size_t)b * ZD + c];
        d_mu [(size_t)b * ZD + c] = mu;
        d_std[(size_t)b * ZD + c] = st;
        d_z  [(size_t)b * (ZD + 2) + c] = z;
        zf[idx] = z;
        z16[idx] = (__bf16)z;
    } else if (c == ZD) {
        float v = rd[b];
        d_z[(size_t)b * (ZD + 2) + c] = v; zf[idx] = v; z16[idx] = (__bf16)v;
    } else if (c == ZD + 1) {
        float v = nd[b];
        d_z[(size_t)b * (ZD + 2) + c] = v; zf[idx] = v; z16[idx] = (__bf16)v;
    } else {
        zf[idx] = 0.0f; z16[idx] = (__bf16)0.0f;
    }
}

// ---------------- decoder input packing ----------------
__global__ void k_prep_dec(const float* __restrict__ x, const float* __restrict__ zf,
                           __bf16* __restrict__ dec) {
    int idx = blockIdx.x * 256 + threadIdx.x;
    if (idx >= TSZ * BSZ * XDEC_P) return;
    int c = idx % XDEC_P;
    int b = (idx / XDEC_P) % BSZ;
    int t = idx / (XDEC_P * BSZ);
    float v;
    if (c < ROLL) {
        v = (t == 0) ? ((c == ROLL - 1) ? 1.0f : 0.0f)
                     : x[((size_t)b * TSZ + (t - 1)) * ROLL + c];
    } else if (c < ROLL + ZD + 2) {
        v = zf[(size_t)b * XENC_P + (c - ROLL)];
    } else v = 0.0f;
    dec[idx] = (__bf16)v;
}

// ---------------- generic casts / zero ----------------
__global__ void k_cast_f32_bf16(const float* __restrict__ in, __bf16* __restrict__ out, int n) {
    int i = blockIdx.x * 256 + threadIdx.x;
    if (i < n) out[i] = (__bf16)in[i];
}
__global__ void k_zero_f32(float* p, int n) {
    int i = blockIdx.x * 256 + threadIdx.x;
    if (i < n) p[i] = 0.0f;
}
__global__ void k_zero_bf16(__bf16* p, int n) {
    int i = blockIdx.x * 256 + threadIdx.x;
    if (i < n) p[i] = (__bf16)0.0f;
}

// ---------------- batched log_softmax over ROLL=130 ----------------
__global__ void k_logsoftmax(const float* __restrict__ logits, float* __restrict__ out) {
    __shared__ float red[256];
    const int row = blockIdx.x;
    const int b = row % BSZ;
    const int t = row / BSZ;
    const int tid = threadIdx.x;
    const float* in = logits + (size_t)row * ROLL;

    float mx = -3.4e38f;
    for (int i = tid; i < ROLL; i += 256) mx = fmaxf(mx, in[i]);
    red[tid] = mx; __syncthreads();
    for (int s = 128; s > 0; s >>= 1) {
        if (tid < s) red[tid] = fmaxf(red[tid], red[tid + s]);
        __syncthreads();
    }
    mx = red[0]; __syncthreads();

    float sm = 0.0f;
    for (int i = tid; i < ROLL; i += 256) sm += __expf(in[i] - mx);
    red[tid] = sm; __syncthreads();
    for (int s = 128; s > 0; s >>= 1) {
        if (tid < s) red[tid] += red[tid + s];
        __syncthreads();
    }
    const float lse = __logf(red[0]) + mx;

    float* op = out + ((size_t)b * TSZ + t) * ROLL;
    for (int i = tid; i < ROLL; i += 256) op[i] = in[i] - lse;
}

// ---------------- host orchestration ----------------
static inline void gemm(const __bf16* A, int lda, const __bf16* Bm, int ldb,
                        const float* bias, float* C, int ldc,
                        int M, int N, int K, hipStream_t s) {
    dim3 g((N + 127) / 128, M / 64);
    k_gemm_bf16<<<g, 256, 0, s>>>(A, lda, Bm, ldb, bias, C, ldc, M, N, K);
}
static inline int cdiv(int a, int b) { return (a + b - 1) / b; }

extern "C" void kernel_launch(void* const* d_in, const int* in_sizes, int n_in,
                              void* d_out, int out_size, void* d_ws, size_t ws_size,
                              hipStream_t stream) {
    (void)in_sizes; (void)n_in; (void)out_size; (void)ws_size;

    const float* x     = (const float*)d_in[0];
    const float* rd    = (const float*)d_in[4];
    const float* nd    = (const float*)d_in[5];
    const float* eps   = (const float*)d_in[6];
    const float* Wih_f = (const float*)d_in[7];
    const float* Whh_f = (const float*)d_in[8];
    const float* bih_f = (const float*)d_in[9];
    const float* bhh_f = (const float*)d_in[10];
    const float* Wih_b = (const float*)d_in[11];
    const float* Whh_b = (const float*)d_in[12];
    const float* bih_b = (const float*)d_in[13];
    const float* bhh_b = (const float*)d_in[14];
    const float* W_mu  = (const float*)d_in[15];
    const float* b_mu  = (const float*)d_in[16];
    const float* W_var = (const float*)d_in[17];
    const float* b_var = (const float*)d_in[18];
    const float* W_init= (const float*)d_in[19];
    const float* b_init= (const float*)d_in[20];
    const float* Wih1  = (const float*)d_in[21];
    const float* Whh1  = (const float*)d_in[22];
    const float* bih1  = (const float*)d_in[23];
    const float* bhh1  = (const float*)d_in[24];
    const float* Wih2  = (const float*)d_in[25];
    const float* Whh2  = (const float*)d_in[26];
    const float* bih2  = (const float*)d_in[27];
    const float* bhh2  = (const float*)d_in[28];
    const float* W_out = (const float*)d_in[29];
    const float* b_out = (const float*)d_in[30];

    float* dout  = (float*)d_out;
    float* d_mu  = dout + (size_t)BSZ * TSZ * ROLL;
    float* d_std = d_mu + (size_t)BSZ * ZD;
    float* d_z   = d_std + (size_t)BSZ * ZD;

    // --- workspace bump allocator ---
    char* p = (char*)d_ws;
    auto alloc = [&](size_t bytes) -> void* {
        void* r = (void*)p;
        p += (bytes + 255) & ~(size_t)255;
        return r;
    };
    __bf16* tWih_f = (__bf16*)alloc((size_t)XENC_P * G3 * 2);
    __bf16* tWhh_f = (__bf16*)alloc((size_t)HID * G3 * 2);
    __bf16* tWih_b = (__bf16*)alloc((size_t)XENC_P * G3 * 2);
    __bf16* tWhh_b = (__bf16*)alloc((size_t)HID * G3 * 2);
    __bf16* tWih1  = (__bf16*)alloc((size_t)XDEC_P * G3 * 2);
    __bf16* tWhh1  = (__bf16*)alloc((size_t)HID * G3 * 2);
    __bf16* tWih2  = (__bf16*)alloc((size_t)HID * G3 * 2);
    __bf16* tWhh2  = (__bf16*)alloc((size_t)HID * G3 * 2);
    __bf16* tWmu   = (__bf16*)alloc((size_t)HCAT * ZD * 2);
    __bf16* tWvar  = (__bf16*)alloc((size_t)HCAT * ZD * 2);
    __bf16* tWinit = (__bf16*)alloc((size_t)XENC_P * HID * 2);
    __bf16* tWout  = (__bf16*)alloc((size_t)HID * NOUT_P * 2);

    float*  gi     = (float*)alloc((size_t)BSZ * G3 * 4);
    float*  gh     = (float*)alloc((size_t)BSZ * G3 * 4);
    float*  h_f    = (float*)alloc((size_t)BSZ * HID * 4);
    float*  h_b    = (float*)alloc((size_t)BSZ * HID * 4);
    float*  h1     = (float*)alloc((size_t)BSZ * HID * 4);
    float*  h2     = (float*)alloc((size_t)BSZ * HID * 4);
    float*  gm     = (float*)alloc((size_t)BSZ * ZD * 4);
    float*  gv     = (float*)alloc((size_t)BSZ * ZD * 4);
    float*  zf     = (float*)alloc((size_t)BSZ * XENC_P * 4);
    float*  logits = (float*)alloc((size_t)TSZ * BSZ * ROLL * 4);

    __bf16* x_in_f = (__bf16*)alloc((size_t)TSZ * BSZ * XENC_P * 2);
    __bf16* x_in_b = (__bf16*)alloc((size_t)TSZ * BSZ * XENC_P * 2);
    __bf16* dec_in = (__bf16*)alloc((size_t)TSZ * BSZ * XDEC_P * 2);
    __bf16* act_hf = (__bf16*)alloc((size_t)BSZ * HID * 2);
    __bf16* act_hb = (__bf16*)alloc((size_t)BSZ * HID * 2);
    __bf16* act_h1 = (__bf16*)alloc((size_t)BSZ * HID * 2);
    __bf16* act_h2 = (__bf16*)alloc((size_t)BSZ * HID * 2);
    __bf16* hcat   = (__bf16*)alloc((size_t)BSZ * HCAT * 2);
    __bf16* z16    = (__bf16*)alloc((size_t)BSZ * XENC_P * 2);
    __bf16* h2_all = (__bf16*)alloc((size_t)TSZ * BSZ * HID * 2);

    // --- 1) pack weights: transpose + cast to bf16 [KP][NP] ---
    auto tc = [&](const float* W, int N, int K, __bf16* out, int KP, int NP) {
        k_transpose_cast<<<cdiv(KP * NP, 256), 256, 0, stream>>>(W, N, K, out, KP, NP);
    };
    tc(Wih_f, G3, ROLL + 2, tWih_f, XENC_P, G3);
    tc(Whh_f, G3, HID,      tWhh_f, HID,    G3);
    tc(Wih_b, G3, ROLL + 2, tWih_b, XENC_P, G3);
    tc(Whh_b, G3, HID,      tWhh_b, HID,    G3);
    tc(Wih1,  G3, ZD + 2 + ROLL, tWih1, XDEC_P, G3);
    tc(Whh1,  G3, HID,      tWhh1, HID,    G3);
    tc(Wih2,  G3, HID,      tWih2, HID,    G3);
    tc(Whh2,  G3, HID,      tWhh2, HID,    G3);
    tc(W_mu,  ZD, HCAT,     tWmu,  HCAT,   ZD);
    tc(W_var, ZD, HCAT,     tWvar, HCAT,   ZD);
    tc(W_init, HID, ROLL,   tWinit, XENC_P, HID);
    tc(W_out, ROLL, HID,    tWout, HID,    NOUT_P);

    // --- 2) encoder input packing + h0 = 0 ---
    k_prep_enc<<<cdiv(TSZ * BSZ * XENC_P, 256), 256, 0, stream>>>(x, rd, nd, x_in_f, x_in_b);
    k_zero_f32 <<<cdiv(BSZ * HID, 256), 256, 0, stream>>>(h_f, BSZ * HID);
    k_zero_f32 <<<cdiv(BSZ * HID, 256), 256, 0, stream>>>(h_b, BSZ * HID);
    k_zero_bf16<<<cdiv(BSZ * HID, 256), 256, 0, stream>>>(act_hf, BSZ * HID);
    k_zero_bf16<<<cdiv(BSZ * HID, 256), 256, 0, stream>>>(act_hb, BSZ * HID);

    // --- 3) encoder recurrences (Whh bf16 = 25 MB, L2-resident across steps) ---
    const int gatesGrid = (BSZ * HID) / 256;
    for (int t = 0; t < TSZ; ++t) {
        gemm(x_in_f + (size_t)t * BSZ * XENC_P, XENC_P, tWih_f, G3, bih_f, gi, G3,
             BSZ, G3, XENC_P, stream);
        gemm(act_hf, HID, tWhh_f, G3, bhh_f, gh, G3, BSZ, G3, HID, stream);
        k_gates<<<gatesGrid, 256, 0, stream>>>(gi, gh, h_f, h_f, act_hf, nullptr);
    }
    for (int t = 0; t < TSZ; ++t) {
        gemm(x_in_b + (size_t)t * BSZ * XENC_P, XENC_P, tWih_b, G3, bih_b, gi, G3,
             BSZ, G3, XENC_P, stream);
        gemm(act_hb, HID, tWhh_b, G3, bhh_b, gh, G3, BSZ, G3, HID, stream);
        k_gates<<<gatesGrid, 256, 0, stream>>>(gi, gh, h_b, h_b, act_hb, nullptr);
    }

    // --- 4) latent: mu, std, z ---
    k_concat_cast<<<cdiv(BSZ * HCAT, 256), 256, 0, stream>>>(h_f, h_b, hcat);
    gemm(hcat, HCAT, tWmu,  ZD, b_mu,  gm, ZD, BSZ, ZD, HCAT, stream);
    gemm(hcat, HCAT, tWvar, ZD, b_var, gv, ZD, BSZ, ZD, HCAT, stream);
    k_finalize_z<<<cdiv(BSZ * XENC_P, 256), 256, 0, stream>>>(
        gm, gv, eps, rd, nd, d_mu, d_std, d_z, zf, z16);

    // --- 5) decoder init: h1 = z_cat @ W_init.T + b_init ---
    gemm(z16, XENC_P, tWinit, HID, b_init, h1, HID, BSZ, HID, XENC_P, stream);
    k_cast_f32_bf16<<<cdiv(BSZ * HID, 256), 256, 0, stream>>>(h1, act_h1, BSZ * HID);
    k_prep_dec<<<cdiv(TSZ * BSZ * XDEC_P, 256), 256, 0, stream>>>(x, zf, dec_in);

    // --- 6) decoder recurrence: two stacked GRU cells ---
    for (int t = 0; t < TSZ; ++t) {
        // cell 1
        gemm(dec_in + (size_t)t * BSZ * XDEC_P, XDEC_P, tWih1, G3, bih1, gi, G3,
             BSZ, G3, XDEC_P, stream);
        gemm(act_h1, HID, tWhh1, G3, bhh1, gh, G3, BSZ, G3, HID, stream);
        k_gates<<<gatesGrid, 256, 0, stream>>>(gi, gh, h1, h1, act_h1, nullptr);
        // cell 2 (t==0: h2_prev = h1)
        gemm(act_h1, HID, tWih2, G3, bih2, gi, G3, BSZ, G3, HID, stream);
        const __bf16* act_prev2 = (t == 0) ? act_h1 : act_h2;
        const float*  hprev2    = (t == 0) ? h1 : h2;
        gemm(act_prev2, HID, tWhh2, G3, bhh2, gh, G3, BSZ, G3, HID, stream);
        k_gates<<<gatesGrid, 256, 0, stream>>>(gi, gh, hprev2, h2, act_h2,
                                               h2_all + (size_t)t * BSZ * HID);
    }

    // --- 7) batched output projection + log_softmax ---
    gemm(h2_all, HID, tWout, NOUT_P, b_out, logits, ROLL, TSZ * BSZ, ROLL, HID, stream);
    k_logsoftmax<<<TSZ * BSZ, 256, 0, stream>>>(logits, dout);
}
